// Model_36344013259223
// MI455X (gfx1250) — matrix operand order repaired and checked by reading, not compiled, not run on hardware
//
#include <hip/hip_runtime.h>
#include <math.h>

#define B_SZ   32
#define T_SZ   512
#define SSLEN  1024
#define DMODEL 256
#define NLAYER 6
#define DINNER 512
#define NSTATE 16
#define DCONV  4
#define DTRANK 16
#define MROWS  (B_SZ * T_SZ)          // 16384
#define XDB    (DTRANK + 2 * NSTATE)  // 48
#define EPSV   1e-5f

// Block-tiled WMMA GEMM geometry
#define BM     128
#define BN     64
#define KSTEP  32

typedef _Float16 half8 __attribute__((ext_vector_type(8)));
typedef _Float16 v16h  __attribute__((ext_vector_type(16)));
typedef float    v8f   __attribute__((ext_vector_type(8)));
typedef unsigned int u32x4 __attribute__((ext_vector_type(4)));
typedef int          i32x4 __attribute__((ext_vector_type(4)));
typedef int          i32x8 __attribute__((ext_vector_type(8)));

// ---------------------------------------------------------------------------
// helpers
// ---------------------------------------------------------------------------
__global__ void f32_to_f16_kernel(const float* __restrict__ in,
                                  _Float16* __restrict__ out, int n) {
  int i = blockIdx.x * blockDim.x + threadIdx.x;
  if (i < n) out[i] = (_Float16)in[i];
}

__global__ void zero_kernel(float* __restrict__ p, int n) {
  int i = blockIdx.x * blockDim.x + threadIdx.x;
  if (i < n) p[i] = 0.0f;
}

__global__ void zero2_kernel(float* __restrict__ p) { p[0] = 0.0f; p[1] = 0.0f; }

__device__ __forceinline__ v16h pack16(half8 lo, half8 hi) {
  v16h v;
#pragma unroll
  for (int i = 0; i < 8; ++i) { v[i] = lo[i]; v[i + 8] = hi[i]; }
  return v;
}

// ---------------------------------------------------------------------------
// TDM: issue a 2D tile load (tileRows x KSTEP f16) from a row-major
// [rowsTotal x kdim] f16 tensor into LDS at byte offset lds_off.
// D# per CDNA5 ISA 8.3/8.4: group0 {count=1, lds_addr, global_addr, type=2},
// group1 {data_size=2B, tensor_dim0=kdim, tensor_dim1=rowsTotal,
//         tile_dim0=KSTEP, tile_dim1=tileRows, tensor_dim0_stride=kdim}.
// This toolchain exposes the 6-arg builtin (g0, g1, g2, g3, g4, cpol).
// ---------------------------------------------------------------------------
__device__ __forceinline__ void tdm_load_tile(unsigned lds_off,
                                              const _Float16* g,
                                              int kdim, int rowsTotal,
                                              int tileRows) {
  unsigned long long ga = (unsigned long long)(uintptr_t)g;
  u32x4 g0 = {};
  g0[0] = 1u;                                   // count = 1 valid descriptor
  g0[1] = lds_off;                              // LDS byte address
  g0[2] = (unsigned)ga;                         // global_addr[31:0]
  g0[3] = ((unsigned)(ga >> 32) & 0x01FFFFFFu) | (2u << 30);  // addr hi | type=2
  i32x8 g1 = {};
  g1[0] = (int)(1u << 16);                      // data_size = 1 -> 2 bytes
  g1[1] = (int)(((unsigned)kdim & 0xFFFFu) << 16);            // tensor_dim0 lo
  g1[2] = (int)((((unsigned)kdim >> 16) & 0xFFFFu) |
                (((unsigned)rowsTotal & 0xFFFFu) << 16));     // dim0 hi | dim1 lo
  g1[3] = (int)((((unsigned)rowsTotal >> 16) & 0xFFFFu) |
                ((unsigned)KSTEP << 16));                     // dim1 hi | tile_dim0
  g1[4] = (int)((unsigned)tileRows & 0xFFFFu);                // tile_dim1 (tile_dim2=0)
  g1[5] = (int)kdim;                                          // tensor_dim0_stride lo
  g1[6] = 0;
  g1[7] = 0;
  i32x4 gz4 = {};
  i32x8 gz8 = {};
  __builtin_amdgcn_tensor_load_to_lds(g0, g1, gz4, gz4, gz8, 0);
}

// ---------------------------------------------------------------------------
// Block-tiled WMMA GEMM with TDM double-buffered LDS staging.
// C[M,N] = A[M,K] * W[N,K]^T (+bias). 256 threads = 8 waves; block tile
// BM x BN = 128x64; wave tile 32x32 = 2x2 wmma accumulators.
// Requires: M%128==0, N%64==0, K%32==0.
// ---------------------------------------------------------------------------
__global__ __launch_bounds__(256)
void gemm_wmma_f16_tdm(const _Float16* __restrict__ A,
                       const _Float16* __restrict__ W,
                       const float* __restrict__ bias, float* __restrict__ C,
                       int M, int N, int K) {
  __shared__ _Float16 lA[2][BM * KSTEP];
  __shared__ _Float16 lB[2][BN * KSTEP];

  const int tid  = threadIdx.x;
  const int lane = tid & 31;
  const int wv   = tid >> 5;     // 0..7
  const int wm   = wv >> 1;      // 0..3 : 32-row strip
  const int wn   = wv & 1;       // 0..1 : 32-col strip

  const int tilesN = N / BN;
  const int bm0 = (blockIdx.x / tilesN) * BM;
  const int bn0 = (blockIdx.x - (blockIdx.x / tilesN) * tilesN) * BN;

  const int r     = lane & 15;
  const int hsel  = lane >> 4;
  const int akoff = hsel << 3;   // A frag: halves 0..7 = K(akoff..akoff+7), 8..15 = +16
  const int bkoff = hsel << 3;   // B frag: halves 0..7 = K(bkoff..bkoff+7), 8..15 = +16

  const bool lead = (wv == 0);
  unsigned laOff[2], lbOff[2];
  laOff[0] = (unsigned)(uintptr_t)(&lA[0][0]);
  laOff[1] = (unsigned)(uintptr_t)(&lA[1][0]);
  lbOff[0] = (unsigned)(uintptr_t)(&lB[0][0]);
  lbOff[1] = (unsigned)(uintptr_t)(&lB[1][0]);

  if (lead) {
    tdm_load_tile(laOff[0], A + (size_t)bm0 * K, K, M, BM);
    tdm_load_tile(lbOff[0], W + (size_t)bn0 * K, K, N, BN);
  }

  v8f acc[2][2] = {};
  const int nstep = K / KSTEP;
  for (int s = 0; s < nstep; ++s) {
    const int cur = s & 1;
    if (lead) {
      if (s + 1 < nstep) {
        const int k1 = (s + 1) * KSTEP;
        tdm_load_tile(laOff[cur ^ 1], A + (size_t)bm0 * K + k1, K, M, BM);
        tdm_load_tile(lbOff[cur ^ 1], W + (size_t)bn0 * K + k1, K, N, BN);
        __builtin_amdgcn_s_wait_tensorcnt((short)2);  // current buffer complete
      } else {
        __builtin_amdgcn_s_wait_tensorcnt((short)0);
      }
    }
    __syncthreads();  // staged K-slice visible to all waves

    const _Float16* aBase = &lA[cur][0];
    const _Float16* bBase = &lB[cur][0];
    v16h af[2], bf[2];
#pragma unroll
    for (int mi = 0; mi < 2; ++mi) {
      const _Float16* ap = aBase + (wm * 32 + mi * 16 + r) * KSTEP + akoff;
      af[mi] = pack16(*(const half8*)ap, *(const half8*)(ap + 16));
    }
#pragma unroll
    for (int ni = 0; ni < 2; ++ni) {
      const _Float16* bp = bBase + (wn * 32 + ni * 16 + r) * KSTEP + bkoff;
      bf[ni] = pack16(*(const half8*)bp, *(const half8*)(bp + 16));
    }
#pragma unroll
    for (int mi = 0; mi < 2; ++mi)
#pragma unroll
      for (int ni = 0; ni < 2; ++ni)
        acc[mi][ni] = __builtin_amdgcn_wmma_f32_16x16x32_f16(
            false, af[mi], false, bf[ni], (short)0, acc[mi][ni], false, false);

    __syncthreads();  // done reading before buffer is re-filled
  }

#pragma unroll
  for (int mi = 0; mi < 2; ++mi) {
#pragma unroll
    for (int ni = 0; ni < 2; ++ni) {
      const int col = bn0 + wn * 32 + ni * 16 + r;
      const int row = bm0 + wm * 32 + mi * 16 + (hsel << 3);
      const float bv = bias ? bias[col] : 0.0f;
      float* cp = C + (size_t)row * N + col;
#pragma unroll
      for (int i = 0; i < 8; ++i) cp[(size_t)i * N] = acc[mi][ni][i] + bv;
    }
  }
}

// ---------------------------------------------------------------------------
// Direct-load WMMA GEMM (wave per 16x16 tile) — used for thin N (x_proj, N=48)
// ---------------------------------------------------------------------------
__global__ __launch_bounds__(256)
void gemm_wmma_f16(const _Float16* __restrict__ A, const _Float16* __restrict__ W,
                   const float* __restrict__ bias, float* __restrict__ C,
                   int M, int N, int K) {
  const int lane   = threadIdx.x & 31;
  const int wave   = blockIdx.x * (blockDim.x >> 5) + (threadIdx.x >> 5);
  const int tilesN = N >> 4;
  const int tm     = wave / tilesN;
  const int tn     = wave - tm * tilesN;
  const int m0     = tm << 4;
  const int n0     = tn << 4;

  const int r     = lane & 15;
  const int hsel  = lane >> 4;
  const int akoff = hsel << 3;
  const int bkoff = hsel << 3;

  const _Float16* ap = A + (size_t)(m0 + r) * K + akoff;
  const _Float16* wp = W + (size_t)(n0 + r) * K + bkoff;

  v8f acc = {};
  for (int k0 = 0; k0 < K; k0 += 32) {
    half8 alo = *(const half8*)(ap);
    half8 ahi = *(const half8*)(ap + 16);
    half8 blo = *(const half8*)(wp);
    half8 bhi = *(const half8*)(wp + 16);
    ap += 32; wp += 32;
    __builtin_prefetch(ap, 0, 0);
    __builtin_prefetch(wp, 0, 0);
    acc = __builtin_amdgcn_wmma_f32_16x16x32_f16(
        false, pack16(alo, ahi), false, pack16(blo, bhi), (short)0, acc,
        false, false);
  }

  const int   cn  = n0 + r;
  const int   cmb = m0 + (hsel << 3);
  const float bv  = bias ? bias[cn] : 0.0f;
  float* cp = C + (size_t)cmb * N + cn;
#pragma unroll
  for (int i = 0; i < 8; ++i) cp[(size_t)i * N] = acc[i] + bv;
}

// ---------------------------------------------------------------------------
// Global batch-norm statistics then apply + SELU
// ---------------------------------------------------------------------------
__global__ __launch_bounds__(256)
void bn_stats_kernel(const float* __restrict__ x, float* __restrict__ stats, int n) {
  __shared__ float s1[256], s2[256];
  float a = 0.0f, b = 0.0f;
  for (int i = blockIdx.x * blockDim.x + threadIdx.x; i < n;
       i += gridDim.x * blockDim.x) {
    float v = x[i];
    a += v; b += v * v;
  }
  s1[threadIdx.x] = a; s2[threadIdx.x] = b;
  __syncthreads();
  for (int s = 128; s > 0; s >>= 1) {
    if ((int)threadIdx.x < s) {
      s1[threadIdx.x] += s1[threadIdx.x + s];
      s2[threadIdx.x] += s2[threadIdx.x + s];
    }
    __syncthreads();
  }
  if (threadIdx.x == 0) {
    atomicAdd(&stats[0], s1[0]);
    atomicAdd(&stats[1], s2[0]);
  }
}

__global__ void bn_selu_kernel(float* __restrict__ h, const float* __restrict__ stats,
                               const float* __restrict__ g, const float* __restrict__ bb,
                               int n) {
  int i = blockIdx.x * blockDim.x + threadIdx.x;
  if (i >= n) return;
  float mu  = stats[0] / (float)n;
  float var = stats[1] / (float)n - mu * mu;
  float v   = g[0] * (h[i] - mu) * rsqrtf(var + EPSV) + bb[0];
  const float sc = 1.0507009873554805f, al = 1.6732632423543772f;
  h[i] = v > 0.0f ? sc * v : sc * al * (expf(v) - 1.0f);
}

// ---------------------------------------------------------------------------
// RMSNorm over last dim (DMODEL), one block per row
// ---------------------------------------------------------------------------
__global__ __launch_bounds__(DMODEL)
void rmsnorm_f16_kernel(const float* __restrict__ in, const float* __restrict__ w,
                        _Float16* __restrict__ out) {
  __shared__ float red[DMODEL];
  int row = blockIdx.x, tid = threadIdx.x;
  float v = in[(size_t)row * DMODEL + tid];
  red[tid] = v * v;
  __syncthreads();
  for (int s = DMODEL / 2; s > 0; s >>= 1) {
    if (tid < s) red[tid] += red[tid + s];
    __syncthreads();
  }
  float sc = rsqrtf(red[0] * (1.0f / DMODEL) + EPSV);
  out[(size_t)row * DMODEL + tid] = (_Float16)(v * sc * w[tid]);
}

__global__ __launch_bounds__(DMODEL)
void rmsnorm_f32_kernel(const float* __restrict__ in, const float* __restrict__ w,
                        float* __restrict__ out) {
  __shared__ float red[DMODEL];
  int row = blockIdx.x, tid = threadIdx.x;
  float v = in[(size_t)row * DMODEL + tid];
  red[tid] = v * v;
  __syncthreads();
  for (int s = DMODEL / 2; s > 0; s >>= 1) {
    if (tid < s) red[tid] += red[tid + s];
    __syncthreads();
  }
  float sc = rsqrtf(red[0] * (1.0f / DMODEL) + EPSV);
  out[(size_t)row * DMODEL + tid] = v * sc * w[tid];
}

// ---------------------------------------------------------------------------
// Depthwise causal conv (DC=4) + SiLU
// ---------------------------------------------------------------------------
__global__ void conv_silu_kernel(const float* __restrict__ xz,
                                 const float* __restrict__ cw,
                                 const float* __restrict__ cb,
                                 float* __restrict__ u, _Float16* __restrict__ uh) {
  int i = blockIdx.x * blockDim.x + threadIdx.x;
  if (i >= MROWS * DINNER) return;
  int m  = i / DINNER;
  int di = i - m * DINNER;
  int t  = m & (T_SZ - 1);
  float acc = cb[di];
#pragma unroll
  for (int k = 0; k < DCONV; ++k) {
    int tt = t + k - (DCONV - 1);
    if (tt >= 0)
      acc += xz[(size_t)(m + k - (DCONV - 1)) * (2 * DINNER) + di] *
             cw[di * DCONV + k];
  }
  float s = acc / (1.0f + expf(-acc));
  u[i]  = s;
  uh[i] = (_Float16)s;
}

// ---------------------------------------------------------------------------
// dt = softplus(xdbl[:, :16] @ dt_w^T + dt_b)  (K=16 -> scalar)
// ---------------------------------------------------------------------------
__global__ void dt_kernel(const float* __restrict__ xdbl, const float* __restrict__ dtw,
                          const float* __restrict__ dtb, float* __restrict__ dt) {
  int i = blockIdx.x * blockDim.x + threadIdx.x;
  if (i >= MROWS * DINNER) return;
  int m = i / DINNER;
  int d = i - m * DINNER;
  const float* xp = xdbl + (size_t)m * XDB;
  float acc = dtb[d];
#pragma unroll
  for (int r = 0; r < DTRANK; ++r) acc += xp[r] * dtw[d * DTRANK + r];
  dt[i] = (acc > 20.0f) ? acc : log1pf(expf(acc));
}

// ---------------------------------------------------------------------------
// Selective scan: thread per (b, di); 16 states in registers, loop over T
// ---------------------------------------------------------------------------
__global__ __launch_bounds__(256)
void scan_kernel(const float* __restrict__ dt, const float* __restrict__ u,
                 const float* __restrict__ xdbl, const float* __restrict__ Alog,
                 float* __restrict__ y) {
  int idx = blockIdx.x * blockDim.x + threadIdx.x;
  if (idx >= B_SZ * DINNER) return;
  int b  = idx / DINNER;
  int di = idx - b * DINNER;
  float a[NSTATE], h[NSTATE];
#pragma unroll
  for (int n = 0; n < NSTATE; ++n) {
    a[n] = -expf(Alog[di * NSTATE + n]);
    h[n] = 0.0f;
  }
  size_t mbase = (size_t)b * T_SZ;
  for (int t = 0; t < T_SZ; ++t) {
    size_t m   = mbase + t;
    float dtv  = dt[m * DINNER + di];
    float uv   = u[m * DINNER + di];
    const float* bc = xdbl + m * XDB;
    float du = dtv * uv, acc = 0.0f;
#pragma unroll
    for (int n = 0; n < NSTATE; ++n) {
      h[n] = expf(dtv * a[n]) * h[n] + du * bc[DTRANK + n];
      acc += h[n] * bc[DTRANK + NSTATE + n];
    }
    y[m * DINNER + di] = acc;
  }
}

// ---------------------------------------------------------------------------
// y = (y + u*D_skip) * silu(z)
// ---------------------------------------------------------------------------
__global__ void gate_kernel(float* __restrict__ y, const float* __restrict__ u,
                            const float* __restrict__ xz, const float* __restrict__ Dskip,
                            _Float16* __restrict__ yh) {
  int i = blockIdx.x * blockDim.x + threadIdx.x;
  if (i >= MROWS * DINNER) return;
  int m  = i / DINNER;
  int di = i - m * DINNER;
  float z   = xz[(size_t)m * (2 * DINNER) + DINNER + di];
  float val = (y[i] + u[i] * Dskip[di]) * (z / (1.0f + expf(-z)));
  y[i]  = val;
  yh[i] = (_Float16)val;
}

__global__ void add_kernel(float* __restrict__ dst, const float* __restrict__ src, int n) {
  int i = blockIdx.x * blockDim.x + threadIdx.x;
  if (i < n) dst[i] += src[i];
}

// ---------------------------------------------------------------------------
extern "C" void kernel_launch(void* const* d_in, const int* in_sizes, int n_in,
                              void* d_out, int out_size, void* d_ws, size_t ws_size,
                              hipStream_t stream) {
  (void)in_sizes; (void)n_in; (void)out_size; (void)ws_size;

  const float* x         = (const float*)d_in[0];
  const float* lp_w      = (const float*)d_in[1];
  const float* lp_b      = (const float*)d_in[2];
  const float* bn_g      = (const float*)d_in[3];
  const float* bn_b      = (const float*)d_in[4];
  const float* in_proj_w = (const float*)d_in[5];
  const float* conv_w    = (const float*)d_in[6];
  const float* conv_b    = (const float*)d_in[7];
  const float* x_proj_w  = (const float*)d_in[8];
  const float* dt_proj_w = (const float*)d_in[9];
  const float* dt_proj_b = (const float*)d_in[10];
  const float* A_log     = (const float*)d_in[11];
  const float* D_skip    = (const float*)d_in[12];
  const float* out_proj_w= (const float*)d_in[13];
  const float* norm_w    = (const float*)d_in[14];
  const float* norm_f_w  = (const float*)d_in[15];

  char* ws = (char*)d_ws;
  size_t off = 0;
  auto alloc = [&](size_t bytes) -> char* {
    char* p = ws + off;
    off += (bytes + 255) & ~(size_t)255;
    return p;
  };

  _Float16* xb    = (_Float16*)alloc((size_t)MROWS * SSLEN * 2);
  _Float16* lpwh  = (_Float16*)alloc((size_t)DMODEL * SSLEN * 2);
  _Float16* inwh  = (_Float16*)alloc((size_t)NLAYER * 2 * DINNER * DMODEL * 2);
  _Float16* xpwh  = (_Float16*)alloc((size_t)NLAYER * XDB * DINNER * 2);
  _Float16* outwh = (_Float16*)alloc((size_t)NLAYER * DMODEL * DINNER * 2);
  float* hidden   = (float*)alloc((size_t)MROWS * DMODEL * 4);
  float* residual = (float*)alloc((size_t)MROWS * DMODEL * 4);
  _Float16* hidh  = (_Float16*)alloc((size_t)MROWS * DMODEL * 2);
  float* xz       = (float*)alloc((size_t)MROWS * 2 * DINNER * 4);
  float* ubuf     = (float*)alloc((size_t)MROWS * DINNER * 4);
  _Float16* uh    = (_Float16*)alloc((size_t)MROWS * DINNER * 2);
  float* xdbl     = (float*)alloc((size_t)MROWS * XDB * 4);
  float* dtbuf    = (float*)alloc((size_t)MROWS * DINNER * 4);
  float* ybuf     = (float*)alloc((size_t)MROWS * DINNER * 4);
  _Float16* yh    = (_Float16*)alloc((size_t)MROWS * DINNER * 2);
  float* stats    = (float*)alloc(2 * sizeof(float));

  auto cvt = [&](const float* src, _Float16* dst, int n) {
    f32_to_f16_kernel<<<(n + 255) / 256, 256, 0, stream>>>(src, dst, n);
  };
  cvt(x, xb, MROWS * SSLEN);
  cvt(lp_w, lpwh, DMODEL * SSLEN);
  cvt(in_proj_w, inwh, NLAYER * 2 * DINNER * DMODEL);
  cvt(x_proj_w, xpwh, NLAYER * XDB * DINNER);
  cvt(out_proj_w, outwh, NLAYER * DMODEL * DINNER);

  // N%64==0 -> TDM block-tiled kernel; thin N -> direct wave-per-tile kernel
  auto gemm = [&](const _Float16* A, const _Float16* W, const float* bias,
                  float* C, int M, int N, int K) {
    if ((N % BN) == 0 && (M % BM) == 0) {
      int blocks = (M / BM) * (N / BN);
      gemm_wmma_f16_tdm<<<blocks, 256, 0, stream>>>(A, W, bias, C, M, N, K);
    } else {
      int waves = (M / 16) * (N / 16);
      gemm_wmma_f16<<<waves / 8, 256, 0, stream>>>(A, W, bias, C, M, N, K);
    }
  };

  const int nMD = MROWS * DMODEL;   // 4,194,304
  const int nMI = MROWS * DINNER;   // 8,388,608

  // feat = x @ lp_w^T + lp_b -> global BN -> SELU
  gemm(xb, lpwh, lp_b, hidden, MROWS, DMODEL, SSLEN);
  zero2_kernel<<<1, 1, 0, stream>>>(stats);
  bn_stats_kernel<<<1024, 256, 0, stream>>>(hidden, stats, nMD);
  bn_selu_kernel<<<nMD / 256, 256, 0, stream>>>(hidden, stats, bn_g, bn_b, nMD);
  zero_kernel<<<nMD / 256, 256, 0, stream>>>(residual, nMD);

  for (int l = 0; l < NLAYER; ++l) {
    add_kernel<<<nMD / 256, 256, 0, stream>>>(residual, hidden, nMD);
    rmsnorm_f16_kernel<<<MROWS, DMODEL, 0, stream>>>(residual,
                                                     norm_w + (size_t)l * DMODEL, hidh);
    gemm(hidh, inwh + (size_t)l * 2 * DINNER * DMODEL, nullptr, xz,
         MROWS, 2 * DINNER, DMODEL);
    conv_silu_kernel<<<nMI / 256, 256, 0, stream>>>(
        xz, conv_w + (size_t)l * DINNER * DCONV, conv_b + (size_t)l * DINNER, ubuf, uh);
    gemm(uh, xpwh + (size_t)l * XDB * DINNER, nullptr, xdbl, MROWS, XDB, DINNER);
    dt_kernel<<<nMI / 256, 256, 0, stream>>>(
        xdbl, dt_proj_w + (size_t)l * DINNER * DTRANK,
        dt_proj_b + (size_t)l * DINNER, dtbuf);
    scan_kernel<<<(B_SZ * DINNER) / 256, 256, 0, stream>>>(
        dtbuf, ubuf, xdbl, A_log + (size_t)l * DINNER * NSTATE, ybuf);
    gate_kernel<<<nMI / 256, 256, 0, stream>>>(ybuf, ubuf, xz,
                                               D_skip + (size_t)l * DINNER, yh);
    gemm(yh, outwh + (size_t)l * DMODEL * DINNER, nullptr, hidden,
         MROWS, DMODEL, DINNER);
  }

  add_kernel<<<nMD / 256, 256, 0, stream>>>(residual, hidden, nMD);
  rmsnorm_f32_kernel<<<MROWS, DMODEL, 0, stream>>>(residual, norm_f_w, (float*)d_out);
}
